// CorrTorch_49787260895396
// MI455X (gfx1250) — matrix operand order repaired and checked by reading, not compiled, not run on hardware
//
#include <hip/hip_runtime.h>

// ---------------------------------------------------------------------------
// Fused correlation-volume + 1x1x1 conv (channel-mix GEMM) for MI455X/gfx1250.
// out[27, D,H,W] = W[27,864] x cv[864, D*H*W] + b,
//   cv[o*32+ch, v] = x1[ch,v] * x2pad[ch, v+off(o)] * (1/sqrt(32))
// bf16 V_WMMA_F32_16X16X32_BF16, f32 accumulate, cv built on the fly from
// LDS-staged bf16 tiles (scale folded into W). Each wave owns TWO N-tiles so
// A (weight) fragments are loaded once per K-chunk for 4 WMMAs (4 independent
// accumulator chains -> no WMMA->WMMA hazard NOPs, ds/wmma ratio 2.0).
// W is pre-converted once (prep kernel) into its padded bf16 LDS image in
// d_ws, then streamed into LDS with GLOBAL_LOAD_ASYNC_TO_LDS_B128 (ASYNCcnt).
// ---------------------------------------------------------------------------

typedef __bf16 bf16_t;
typedef __attribute__((ext_vector_type(16))) __bf16 v16bf;
typedef __attribute__((ext_vector_type(8)))  float  v8f;

#define C_CH  32
#define DD    64
#define HH    64
#define WW    64
#define NOFF  27
#define KTOT  864          // 27 * 32
#define KPAD  872          // 16B-aligned bf16 rows (1744 B), good bank spread
#define VOXEL (DD*HH*WW)   // 262144
#define WBYTES (32 * KPAD * 2)   // 55808: padded bf16 W image

// LDS layout (bytes), all 16B aligned:
//   W tiles : 32 rows x KPAD bf16                = 55808
//   x1 tile : 128 voxels x 80 B (32ch bf16+pad)  = 10240
//   x2 halo : 3*4*66 cells x 80 B                = 63360
#define W_OFF   0
#define X1_OFF  55808
#define X2_OFF  (55808 + 10240)
#define SMEM_BYTES (55808 + 10240 + 63360)   // 129408 -> 2 blocks per WGP

#define NTHREADS 128   // 4 wave32s; each wave handles 2 N-tiles (32 voxels)

union FragBF {
    uint4 q[2];
    v16bf v;
};

// ---- prep: conv_w -> padded bf16 LDS image (scale folded, zero padding) ----
__global__ void prep_w_bf16(const float* __restrict__ conv_w,
                            bf16_t* __restrict__ wbf) {
    const int idx = blockIdx.x * 256 + threadIdx.x;
    if (idx >= 32 * KPAD) return;
    const int m = idx / KPAD;
    const int k = idx - m * KPAD;
    const float scale = 0.17677669529663687f;  // 1/sqrt(C)
    const float v = (m < NOFF && k < KTOT) ? conv_w[m * KTOT + k] * scale : 0.0f;
    wbf[idx] = (bf16_t)v;
}

__global__ void corr_fused_wmma(const float* __restrict__ in1,
                                const float* __restrict__ in2,
                                const bf16_t* __restrict__ wbf,
                                const float* __restrict__ conv_b,
                                float* __restrict__ out) {
    extern __shared__ char smem[];
    bf16_t* wl  = (bf16_t*)(smem + W_OFF);
    char*   x1l = smem + X1_OFF;
    char*   x2l = smem + X2_OFF;

    const int tid = threadIdx.x;
    const int bid = blockIdx.x;
    const int d0  = bid >> 5;         // 64 depth slices
    const int h0  = (bid & 31) << 1;  // 32 pairs of h-rows

    // ---- stage W: async memory->LDS copy of the prebuilt bf16 image ----
    {
        const char* src = (const char*)wbf;
        for (int off = tid * 16; off < WBYTES; off += NTHREADS * 16) {
            unsigned long long ga = (unsigned long long)(src + off);
            unsigned int       la = (unsigned int)(W_OFF + off);
            asm volatile("global_load_async_to_lds_b128 %0, %1, off"
                         :: "v"(la), "v"(ga) : "memory");
        }
    }

    // ---- stage x1 tile: (h0..h0+1, all w) at depth d0, ch-contiguous ----
    for (int idx = tid; idx < 2 * WW * C_CH; idx += NTHREADS) {
        const int w    = idx & 63;
        const int rest = idx >> 6;
        const int hrow = rest & 1;
        const int ch   = rest >> 1;
        const float v  = in1[((ch * DD + d0) * HH + (h0 + hrow)) * WW + w];
        *(bf16_t*)(x1l + (hrow * WW + w) * 80 + ch * 2) = (bf16_t)v;
    }

    // ---- stage x2 halo: [3 dz][4 hy][66 wx][32 ch], zero-padded OOB ----
    for (int idx = tid; idx < 3 * 4 * 66 * C_CH; idx += NTHREADS) {
        const int wi = idx % 66;
        int rest     = idx / 66;
        const int yi = rest & 3;  rest >>= 2;
        const int zi = rest % 3;
        const int ch = rest / 3;
        const int dg = d0 - 1 + zi;
        const int hg = h0 - 1 + yi;
        const int wg = wi - 1;
        float v = 0.0f;
        if ((unsigned)dg < DD && (unsigned)hg < HH && (unsigned)wg < WW)
            v = in2[((ch * DD + dg) * HH + hg) * WW + wg];
        *(bf16_t*)(x2l + ((zi * 4 + yi) * 66 + wi) * 80 + ch * 2) = (bf16_t)v;
    }

    // async W copy must land before the barrier releases consumers
    asm volatile("s_wait_asynccnt 0x0" ::: "memory");
    __syncthreads();

    // ---- per-wave GEMM: two N-tiles of 16 voxels, M = 32 (two 16-row tiles)
    const int lane   = tid & 31;
    const int wv     = tid >> 5;       // 4 waves, each owns 32 voxels
    const int n      = lane & 15;      // N index == A-matrix row index
    const int chHalf = lane >> 4;      // selects ch 0..15 vs 16..31 (B K-half)
    const int tA     = wv * 32 + n;    // voxel for N-tile A
    const int hrow   = wv >> 1;        // waves 0,1 -> row 0; 2,3 -> row 1
    const int wA     = tA & 63;
    const int wB     = wA + 16;        // voxel for N-tile B (tA + 16)

    // x1 B-fragment halves: invariant across the 27 offsets
    FragBF x1A, x1B;
    {
        const char* pA = x1l + (hrow * WW + wA) * 80 + chHalf * 32;
        x1A.q[0] = *(const uint4*)(pA);
        x1A.q[1] = *(const uint4*)(pA + 16);
        const char* pB = x1l + (hrow * WW + wB) * 80 + chHalf * 32;
        x1B.q[0] = *(const uint4*)(pB);
        x1B.q[1] = *(const uint4*)(pB + 16);
    }

    v8f acc00 = {};  // M-tile 0 (o 0..15)  x N-tile A
    v8f acc01 = {};  // M-tile 0            x N-tile B
    v8f acc10 = {};  // M-tile 1 (o 16..31) x N-tile A
    v8f acc11 = {};  // M-tile 1            x N-tile B

#pragma unroll
    for (int dz = 0; dz < 3; ++dz) {
#pragma unroll
        for (int dy = 0; dy < 3; ++dy) {
#pragma unroll
            for (int dx = 0; dx < 3; ++dx) {
                const int op = (dz * 3 + dy) * 3 + dx;   // K-chunk index
                const int rowOff = (dz * 4 + (hrow + dy)) * 66;

                // B fragments: cv chunk = bf16(x1) * bf16(x2 shifted)
                FragBF x2A, x2B, af0, af1;
                const char* p2A = x2l + (rowOff + (wA + dx)) * 80 + chHalf * 32;
                x2A.q[0] = *(const uint4*)(p2A);
                x2A.q[1] = *(const uint4*)(p2A + 16);
                const char* p2B = x2l + (rowOff + (wB + dx)) * 80 + chHalf * 32;
                x2B.q[0] = *(const uint4*)(p2B);
                x2B.q[1] = *(const uint4*)(p2B + 16);
                v16bf bvA = x1A.v * x2A.v;  // packed bf16 multiply
                v16bf bvB = x1B.v * x2B.v;

                // A fragments: shared by both N-tiles -> loaded once per chunk
                const char* pw = (const char*)wl +
                    n * (KPAD * 2) + op * 64 + chHalf * 32;
                af0.q[0] = *(const uint4*)(pw);
                af0.q[1] = *(const uint4*)(pw + 16);
                const char* pw1 = pw + 16 * (KPAD * 2);
                af1.q[0] = *(const uint4*)(pw1);
                af1.q[1] = *(const uint4*)(pw1 + 16);

                // 4 independent accumulator chains
                acc00 = __builtin_amdgcn_wmma_f32_16x16x32_bf16(
                    false, af0.v, false, bvA, (short)0, acc00, false, false);
                acc01 = __builtin_amdgcn_wmma_f32_16x16x32_bf16(
                    false, af0.v, false, bvB, (short)0, acc01, false, false);
                acc10 = __builtin_amdgcn_wmma_f32_16x16x32_bf16(
                    false, af1.v, false, bvA, (short)0, acc10, false, false);
                acc11 = __builtin_amdgcn_wmma_f32_16x16x32_bf16(
                    false, af1.v, false, bvB, (short)0, acc11, false, false);
            }
        }
    }

    // ---- epilogue: bias add + store (C/D layout: VGPR r -> M = r + 8*chHalf)
    const int rowBase = (d0 * HH + (h0 + hrow)) * WW;
    const int sA = rowBase + wA;
    const int sB = rowBase + wB;
#pragma unroll
    for (int r = 0; r < 8; ++r) {
        const int o0 = chHalf * 8 + r;        // 0..15, always valid
        const float b0 = conv_b[o0];
        out[o0 * VOXEL + sA] = acc00[r] + b0;
        out[o0 * VOXEL + sB] = acc01[r] + b0;
        const int o1 = 16 + chHalf * 8 + r;   // 16..31, valid if < 27
        if (o1 < NOFF) {
            const float b1 = conv_b[o1];
            out[o1 * VOXEL + sA] = acc10[r] + b1;
            out[o1 * VOXEL + sB] = acc11[r] + b1;
        }
    }
}

extern "C" void kernel_launch(void* const* d_in, const int* in_sizes, int n_in,
                              void* d_out, int out_size, void* d_ws, size_t ws_size,
                              hipStream_t stream) {
    (void)in_sizes; (void)n_in; (void)out_size; (void)ws_size;
    const float* in1 = (const float*)d_in[0];
    const float* in2 = (const float*)d_in[1];
    const float* cw  = (const float*)d_in[2];
    const float* cb  = (const float*)d_in[3];
    float* out   = (float*)d_out;
    bf16_t* wbf  = (bf16_t*)d_ws;    // 55808 B of scratch for the W image

    // Allow >48KB dynamic LDS (idempotent, deterministic, not a stream op).
    (void)hipFuncSetAttribute((const void*)corr_fused_wmma,
                              hipFuncAttributeMaxDynamicSharedMemorySize,
                              SMEM_BYTES);

    // 1) build padded bf16 W image once per call
    prep_w_bf16<<<dim3((32 * KPAD + 255) / 256), dim3(256), 0, stream>>>(cw, wbf);

    // 2) fused correlation + GEMM
    const dim3 grid(DD * (HH / 2));  // 2048 blocks
    const dim3 block(NTHREADS);      // 4 wave32s
    corr_fused_wmma<<<grid, block, SMEM_BYTES, stream>>>(in1, in2, wbf, cb, out);
}
